// WindowAttention_60344290509066
// MI455X (gfx1250) — matrix operand order repaired and checked by reading, not compiled, not run on hardware
//
#include <hip/hip_runtime.h>
#include <hip/hip_bf16.h>

#define DIM   384
#define HEADS 12
#define NTOK  49
#define DHEAD 32
#define SCALE 0.17677669529663687f  /* 32^-0.5 */

typedef __attribute__((ext_vector_type(16))) __bf16        v16bf;
typedef __attribute__((ext_vector_type(8)))  float         v8f;
typedef __attribute__((ext_vector_type(8)))  unsigned int  v8u;

__device__ __forceinline__ unsigned short f2bf(float f) {
  unsigned u = __float_as_uint(f);
  return (unsigned short)((u + 0x7FFFu + ((u >> 16) & 1u)) >> 16);  // RNE
}

__device__ __forceinline__ v16bf mkfrag(uint4 a, uint4 b) {
  v8u r = {a.x, a.y, a.z, a.w, b.x, b.y, b.z, b.w};
  return __builtin_bit_cast(v16bf, r);
}

// A fragment (16x32 bf16) from row-major [M][K] storage.
// lanes 0-15: M=lane, K = 0..7 & 16..23 ; lanes 16-31: M=lane-16, K = 8..15 & 24..31
__device__ __forceinline__ v16bf fragA(const unsigned short* p, int stride,
                                       int m0, int k0, int lane) {
  int m  = m0 + (lane & 15);
  int kb = k0 + ((lane >> 4) << 3);
  const unsigned short* q = p + m * stride + kb;
  return mkfrag(*(const uint4*)q, *(const uint4*)(q + 16));
}

// B fragment (32x16 bf16) from [N][K] (transposed) storage:
// lanes 0-15: N=lane, K = 0..7 & 16..23 ; lanes 16-31: N=lane-16, K = 8..15 & 24..31
__device__ __forceinline__ v16bf fragB(const unsigned short* p, int stride,
                                       int n0, int k0, int lane) {
  int n  = n0 + (lane & 15);
  int kb = k0 + ((lane >> 4) << 3);
  const unsigned short* q = p + n * stride + kb;
  return mkfrag(*(const uint4*)q, *(const uint4*)(q + 16));
}

// ---------------------------------------------------------------------------
// Prep: bf16-transpose weights, pre-gather relative-position bias per head.
// ---------------------------------------------------------------------------
__global__ void swin_prep_kernel(const float* __restrict__ qkv_w,
                                 const float* __restrict__ out_w,
                                 const float* __restrict__ table,
                                 const int*   __restrict__ ridx,
                                 unsigned short* __restrict__ wt_qkv,
                                 unsigned short* __restrict__ wt_out,
                                 float* __restrict__ biasF) {
  int i = blockIdx.x * blockDim.x + threadIdx.x;
  if (i < 3 * DIM * DIM) {                    // wt_qkv[n][k] = bf16(qkv_w[k][n])
    int n = i / DIM, k = i % DIM;
    wt_qkv[i] = f2bf(qkv_w[k * (3 * DIM) + n]);
    return;
  }
  int j = i - 3 * DIM * DIM;
  if (j < DIM * DIM) {                        // wt_out[n][k] = bf16(out_w[k][n])
    int n = j / DIM, k = j % DIM;
    wt_out[j] = f2bf(out_w[k * DIM + n]);
    return;
  }
  int l = j - DIM * DIM;
  if (l < HEADS * NTOK * NTOK) {              // biasF[h][q][j] = table[idx[q][j]][h]
    int h = l / (NTOK * NTOK);
    int r = l % (NTOK * NTOK);
    biasF[l] = table[ridx[r] * HEADS + h];
  }
}

// ---------------------------------------------------------------------------
// Fully-fused window attention: one workgroup (8 wave32) per window.
// ---------------------------------------------------------------------------
__global__ __launch_bounds__(256)
void swin_attn_kernel(const float* __restrict__ x,
                      const float* __restrict__ maskp,
                      const float* __restrict__ qkv_b,
                      const float* __restrict__ out_b,
                      const unsigned short* __restrict__ wt_qkv,
                      const unsigned short* __restrict__ wt_out,
                      const float* __restrict__ biasF,
                      float* __restrict__ out, int nW) {
  extern __shared__ char smraw[];
  unsigned short* sX  = (unsigned short*)smraw;     // [64][384] bf16  49152 B
  unsigned short* sAT = sX  + 64 * DIM;             // [64][384] bf16  49152 B
  unsigned short* sQ  = sAT + 64 * DIM;             // [64][32]  bf16   4096 B
  unsigned short* sK  = sQ  + 64 * DHEAD;           // [64][32]  bf16   4096 B
  unsigned short* sVt = sK  + 64 * DHEAD;           // [32][64]  bf16   4096 B (V^T)
  unsigned short* sP  = sVt + DHEAD * 64;           // [64][64]  bf16   8192 B
  float*          sS   = (float*)(sP + 64 * 64);    // [64][64]  f32   16384 B
  float*          sInv = sS + 64 * 64;              // [64]      f32     256 B

  const int b    = blockIdx.x;
  const int tid  = threadIdx.x;
  const int wave = tid >> 5;
  const int lane = tid & 31;
  const int lo   = lane & 15, hi = lane >> 4;
  const int mb   = b % nW;

  // ---- stage X (fp32 -> bf16), zero pad rows 49..63 ----
  const float* xb = x + (size_t)b * NTOK * DIM;
  for (int i = tid; i < NTOK * DIM; i += 256) sX[i] = f2bf(xb[i]);
  for (int i = tid; i < (64 - NTOK) * DIM; i += 256) sX[NTOK * DIM + i] = 0;
  __syncthreads();

  for (int h = 0; h < HEADS; ++h) {
    // ---- phase 1: Q (scaled), K, V^T for head h (24 tiles, 3 per wave) ----
    for (int i = 0; i < 3; ++i) {
      int t   = wave * 3 + i;
      int mat = t >> 3;                 // 0=Q 1=K 2=V
      int mt  = (t & 7) >> 1;
      int nt  = t & 1;
      int colBase = mat * DIM + h * DHEAD + nt * 16;
      v8f acc = {};
      for (int ks = 0; ks < DIM / 32; ++ks) {
        v16bf a  = fragA(sX, DIM, mt * 16, ks * 32, lane);
        v16bf bm = fragB(wt_qkv, DIM, colBase, ks * 32, lane);
        acc = __builtin_amdgcn_wmma_f32_16x16x32_bf16(false, a, false, bm,
                                                      (short)0, acc, false, false);
      }
      if (mat == 2) {                   // V stored transposed -> B-matrix layout
        int dcol = nt * 16 + lo;
        float bias = qkv_b[2 * DIM + h * DHEAD + dcol];
        unsigned p0 = f2bf(acc[0] + bias) | ((unsigned)f2bf(acc[1] + bias) << 16);
        unsigned p1 = f2bf(acc[2] + bias) | ((unsigned)f2bf(acc[3] + bias) << 16);
        unsigned p2 = f2bf(acc[4] + bias) | ((unsigned)f2bf(acc[5] + bias) << 16);
        unsigned p3 = f2bf(acc[6] + bias) | ((unsigned)f2bf(acc[7] + bias) << 16);
        uint4 val; val.x = p0; val.y = p1; val.z = p2; val.w = p3;
        *(uint4*)(sVt + dcol * 64 + mt * 16 + hi * 8) = val;
      } else {
        unsigned short* dst = mat ? sK : sQ;
        int col = nt * 16 + lo;
        float bias = qkv_b[mat * DIM + h * DHEAD + col];
        float sc   = mat ? 1.0f : SCALE;      // fold softmax scale into Q
        for (int r = 0; r < 8; ++r) {
          int m = mt * 16 + hi * 8 + r;
          dst[m * DHEAD + col] = f2bf((acc[r] + bias) * sc);
        }
      }
    }
    __syncthreads();

    // ---- phase 2: S = Q K^T (+rel-bias +mask at store), 16 tiles ----
    const float* bF = biasF + h * NTOK * NTOK;
    const float* mF = maskp + (size_t)mb * NTOK * NTOK;
    for (int i = 0; i < 2; ++i) {
      int t  = wave * 2 + i;
      int mt = t >> 2, nt = t & 3;
      v8f acc = {};
      v16bf a  = fragA(sQ, DHEAD, mt * 16, 0, lane);
      v16bf bm = fragB(sK, DHEAD, nt * 16, 0, lane);
      acc = __builtin_amdgcn_wmma_f32_16x16x32_bf16(false, a, false, bm,
                                                    (short)0, acc, false, false);
      int j = nt * 16 + lo;
      for (int r = 0; r < 8; ++r) {
        int m = mt * 16 + hi * 8 + r;
        float v = acc[r];
        if (m < NTOK && j < NTOK) v += bF[m * NTOK + j] + mF[m * NTOK + j];
        sS[m * 64 + j] = v;
      }
    }
    __syncthreads();

    // ---- phase 3: row softmax, unnormalized exp; keep 1/rowsum ----
    if (tid < 64) {
      if (tid < NTOK) {
        const float* row = sS + tid * 64;
        float mx = -3.0e38f;
        for (int j = 0; j < NTOK; ++j) mx = fmaxf(mx, row[j]);
        float sum = 0.0f;
        for (int j = 0; j < NTOK; ++j) {
          float e = __expf(row[j] - mx);
          sum += e;
          sP[tid * 64 + j] = f2bf(e);
        }
        for (int j = NTOK; j < 64; ++j) sP[tid * 64 + j] = 0;
        sInv[tid] = 1.0f / sum;
      } else {
        for (int j = 0; j < 64; ++j) sP[tid * 64 + j] = 0;
        sInv[tid] = 0.0f;
      }
    }
    __syncthreads();

    // ---- phase 4: O = P V  (8 tiles, one per wave); normalize at store ----
    {
      int mt = wave >> 1, nt = wave & 1;
      v8f acc = {};
      for (int ks = 0; ks < 2; ++ks) {
        v16bf a  = fragA(sP, 64, mt * 16, ks * 32, lane);
        v16bf bm = fragB(sVt, 64, nt * 16, ks * 32, lane);
        acc = __builtin_amdgcn_wmma_f32_16x16x32_bf16(false, a, false, bm,
                                                      (short)0, acc, false, false);
      }
      int col = h * DHEAD + nt * 16 + lo;
      for (int r = 0; r < 8; ++r) {
        int m = mt * 16 + hi * 8 + r;
        sAT[m * DIM + col] = f2bf(acc[r] * sInv[m]);
      }
    }
    __syncthreads();
  }

  // ---- final projection: out = AT @ out_w + out_b  (96 tiles, 12 per wave) ----
  float* ob = out + (size_t)b * NTOK * DIM;
  for (int i = 0; i < 12; ++i) {
    int t  = wave * 12 + i;
    int mt = t & 3, nt = t >> 2;
    v8f acc = {};
    for (int ks = 0; ks < DIM / 32; ++ks) {
      v16bf a  = fragA(sAT, DIM, mt * 16, ks * 32, lane);
      v16bf bm = fragB(wt_out, DIM, nt * 16, ks * 32, lane);
      acc = __builtin_amdgcn_wmma_f32_16x16x32_bf16(false, a, false, bm,
                                                    (short)0, acc, false, false);
    }
    int g = nt * 16 + lo;
    float bias = out_b[g];
    for (int r = 0; r < 8; ++r) {
      int m = mt * 16 + hi * 8 + r;
      if (m < NTOK) ob[m * DIM + g] = acc[r] + bias;
    }
  }
}

// ---------------------------------------------------------------------------
extern "C" void kernel_launch(void* const* d_in, const int* in_sizes, int n_in,
                              void* d_out, int out_size, void* d_ws, size_t ws_size,
                              hipStream_t stream) {
  const float* x      = (const float*)d_in[0];
  const float* maskp  = (const float*)d_in[1];
  const float* qkv_w  = (const float*)d_in[2];
  const float* qkv_b  = (const float*)d_in[3];
  const float* table  = (const float*)d_in[4];
  const int*   ridx   = (const int*)  d_in[5];
  const float* out_w  = (const float*)d_in[6];
  const float* out_b  = (const float*)d_in[7];
  float*       out    = (float*)d_out;

  char* ws = (char*)d_ws;
  unsigned short* wt_qkv = (unsigned short*)ws;                               // 884736 B
  unsigned short* wt_out = (unsigned short*)(ws + 3 * DIM * DIM * 2);         // 294912 B
  float*          biasF  = (float*)(ws + 3 * DIM * DIM * 2 + DIM * DIM * 2);  // 115248 B

  int Bn = in_sizes[0] / (NTOK * DIM);      // 4096 windows
  int nW = in_sizes[1] / (NTOK * NTOK);     // 64 mask groups

  int prepN = 3 * DIM * DIM + DIM * DIM + HEADS * NTOK * NTOK;
  swin_prep_kernel<<<(prepN + 255) / 256, 256, 0, stream>>>(
      qkv_w, out_w, table, ridx, wt_qkv, wt_out, biasF);

  const int SMEM = 2 * (64 * DIM) * 2          // sX + sAT
                 + 2 * (64 * DHEAD) * 2        // sQ + sK
                 + (DHEAD * 64) * 2            // sVt
                 + (64 * 64) * 2               // sP
                 + (64 * 64) * 4               // sS
                 + 64 * 4;                     // sInv  => 135424 B
  hipFuncSetAttribute(reinterpret_cast<const void*>(swin_attn_kernel),
                      hipFuncAttributeMaxDynamicSharedMemorySize, SMEM);
  swin_attn_kernel<<<Bn, 256, SMEM, stream>>>(
      x, maskp, qkv_b, out_b, wt_qkv, wt_out, biasF, out, nW);
}
